// Bi_lstm_46780783788462
// MI455X (gfx1250) — matrix operand order repaired and checked by reading, not compiled, not run on hardware
//
#include <hip/hip_runtime.h>

typedef __attribute__((ext_vector_type(16))) _Float16 v16h;
typedef __attribute__((ext_vector_type(8)))  float    v8f;

#define H   32
#define SEQ 1024
#define NI  3
#define GSTRIDE 132   // padded row stride for gate buffer (bank-conflict avoidance)

// CDNA5 has a native TRANS32 tanh (v_tanh_f32). Not exposed as a builtin, so use
// inline asm; the trailing v_nop satisfies the trans-op RAW hazard rule (1
// instruction gap before the result may be consumed), since the compiler cannot
// see hazards inside an asm block.
__device__ __forceinline__ float fast_tanh(float x) {
    float r;
    asm("v_tanh_f32 %0, %1\n\tv_nop" : "=v"(r) : "v"(x));
    return r;
}
__device__ __forceinline__ float fast_sigmoid(float x) {
    return __builtin_fmaf(0.5f, fast_tanh(0.5f * x), 0.5f);
}

// One block = 8 waves = 256 threads, handles 16 batch rows for the whole sequence.
// Wave w computes gate columns [16w, 16w+16) via WMMA; each wave then owns the
// elementwise LSTM update for rows (w) and (w+8), lane == hidden index.
__global__ __launch_bounds__(256)
void lstm_attn_fused_kernel(const float* __restrict__ x,
                            const float* __restrict__ W_ih,
                            const float* __restrict__ W_hh,
                            const float* __restrict__ b_ih,
                            const float* __restrict__ b_hh,
                            const float* __restrict__ attn_w,
                            const float* __restrict__ fc1_w,
                            const float* __restrict__ fc1_b,
                            const float* __restrict__ fc2_w,
                            const float* __restrict__ fc2_b,
                            float* __restrict__ out)
{
    __shared__ _Float16 hf16[16 * H];          // h tile (f16) = next WMMA A operand
    __shared__ float    gbuf[16 * GSTRIDE];    // gate pre-activations [16][128] (padded)
    __shared__ float    xbuf[16 * NI];         // x[:, t, :] tile
    __shared__ float    pbuf[16 * (H + 1)];    // pooled vectors for the FC head

    const int tid  = threadIdx.x;
    const int lane = tid & 31;
    const int wave = tid >> 5;                 // 0..7
    const int r0   = blockIdx.x * 16;          // first batch row of this tile

    // ---- loop-invariant B operand: W_hh^T tile for gates n = 16*wave + (lane&15)
    // B (32x16) k order used here (contiguous, same as A): lanes 0-15 hold K=0..15, lanes 16-31 hold K=16..31,
    // two K-values per VGPR ascending.
    const int nB    = wave * 16 + (lane & 15);
    const int kbase = (lane >= 16) ? 16 : 0;
    v16h bmat;
    #pragma unroll
    for (int j = 0; j < 16; ++j)
        bmat[j] = (_Float16)W_hh[nB * H + kbase + j];

    // per-thread input-projection coefficients for gate rows n = q*32 + lane
    float wx0[4], wx1[4], wx2[4], bg[4];
    #pragma unroll
    for (int q = 0; q < 4; ++q) {
        int n = q * H + lane;
        wx0[q] = W_ih[n * NI + 0];
        wx1[q] = W_ih[n * NI + 1];
        wx2[q] = W_ih[n * NI + 2];
        bg[q]  = b_ih[n] + b_hh[n];
    }
    const float aw = attn_w[lane];

    // LSTM cell state + online-softmax attention state for rows (wave) and (wave+8)
    float cst[2]  = {0.0f, 0.0f};
    float pacc[2] = {0.0f, 0.0f};      // running weighted-h accumulator
    float mrow[2] = {-1e30f, -1e30f};  // running max of relu(score)
    float drow[2] = {0.0f, 0.0f};      // running softmax denominator

    for (int i = tid; i < 16 * H; i += 256) hf16[i] = (_Float16)0.0f;
    __syncthreads();

    // WMMA C-layout coordinates for this lane (16x16 f32, 8 VGPRs)
    const int cM   = (lane >= 16) ? 8 : 0;
    const int cN   = lane & 15;
    // A-operand (16-bit 16x32) half-index base: row M = lane&15;
    // lanes 0-15 take K 0..15, lanes 16-31 take K 16..31 (matches B)
    const int aOff = (lane & 15) * H + ((lane >= 16) ? 16 : 0);

    for (int t = 0; t < SEQ; ++t) {
        // stage x[:, t, :] (48 scalars) and prefetch the next timestep
        if (tid < 16 * NI) {
            int r = tid / NI, i = tid % NI;
            xbuf[tid] = x[((size_t)(r0 + r) * SEQ + t) * NI + i];
            if (t + 1 < SEQ)
                __builtin_prefetch(&x[((size_t)(r0 + r) * SEQ + t + 1) * NI + i], 0, 3);
        }

        // load A operand (h tile, f16) from LDS per the WMMA A layout
        v16h amat;
        #pragma unroll
        for (int j = 0; j < 8; ++j) {
            amat[j]     = hf16[aOff + j];
            amat[8 + j] = hf16[aOff + 8 + j];
        }

        // g_raw = h_t @ W_hh^T  (this wave's 16 gate columns)
        v8f cacc = {};
        cacc = __builtin_amdgcn_wmma_f32_16x16x32_f16(
            false, amat, false, bmat, (short)0, cacc, false, false);

        // scatter C into the shared gate buffer
        #pragma unroll
        for (int k = 0; k < 8; ++k)
            gbuf[(cM + k) * GSTRIDE + wave * 16 + cN] = cacc[k];

        __syncthreads();   // gbuf + xbuf visible to all waves

        #pragma unroll
        for (int e = 0; e < 2; ++e) {
            int r = wave + 8 * e;
            float x0 = xbuf[r * NI + 0];
            float x1 = xbuf[r * NI + 1];
            float x2 = xbuf[r * NI + 2];
            float gq[4];
            #pragma unroll
            for (int q = 0; q < 4; ++q)
                gq[q] = gbuf[r * GSTRIDE + q * H + lane]
                      + bg[q] + wx0[q] * x0 + wx1[q] * x1 + wx2[q] * x2;

            float ig = fast_sigmoid(gq[0]);
            float fg = fast_sigmoid(gq[1]);
            float gg = fast_tanh(gq[2]);
            float og = fast_sigmoid(gq[3]);
            cst[e] = fg * cst[e] + ig * gg;
            float hh = og * fast_tanh(cst[e]);

            // attention score: reduce hh*attn_w over the hidden dim (32 lanes)
            float sv = hh * aw;
            #pragma unroll
            for (int ofs = 16; ofs > 0; ofs >>= 1)
                sv += __shfl_xor(sv, ofs, 32);

            // online softmax of relu(score): rescale running accumulators
            float sr = fmaxf(sv, 0.0f);
            float mn = fmaxf(mrow[e], sr);
            float sc = __expf(mrow[e] - mn);
            float wt = __expf(sr - mn);
            drow[e]  = drow[e] * sc + wt;
            pacc[e]  = pacc[e] * sc + wt * hh;
            mrow[e]  = mn;

            hf16[r * H + lane] = (_Float16)hh;   // stage h_{t+1} A operand
        }
        __syncthreads();   // hf16 writes done before next iteration's A loads
    }

    // pooled = pacc / denom
    #pragma unroll
    for (int e = 0; e < 2; ++e)
        pbuf[(wave + 8 * e) * (H + 1) + lane] = pacc[e] / drow[e];
    __syncthreads();

    // tiny FC head: 32 -> 16 (relu) -> 2 ; one thread per batch row
    if (tid < 16) {
        int r = tid;
        float h1[16];
        #pragma unroll
        for (int j = 0; j < 16; ++j) {
            float acc = fc1_b[j];
            #pragma unroll
            for (int h = 0; h < H; ++h)
                acc += pbuf[r * (H + 1) + h] * fc1_w[j * H + h];
            h1[j] = fmaxf(acc, 0.0f);
        }
        #pragma unroll
        for (int k = 0; k < 2; ++k) {
            float acc = fc2_b[k];
            #pragma unroll
            for (int j = 0; j < 16; ++j)
                acc += h1[j] * fc2_w[k * 16 + j];
            out[(size_t)(r0 + r) * 2 + k] = acc;
        }
    }
}

extern "C" void kernel_launch(void* const* d_in, const int* in_sizes, int n_in,
                              void* d_out, int out_size, void* d_ws, size_t ws_size,
                              hipStream_t stream) {
    const float* x      = (const float*)d_in[0];
    const float* W_ih   = (const float*)d_in[1];
    const float* W_hh   = (const float*)d_in[2];
    const float* b_ih   = (const float*)d_in[3];
    const float* b_hh   = (const float*)d_in[4];
    const float* attn_w = (const float*)d_in[5];
    const float* fc1_w  = (const float*)d_in[6];
    const float* fc1_b  = (const float*)d_in[7];
    const float* fc2_w  = (const float*)d_in[8];
    const float* fc2_b  = (const float*)d_in[9];
    float* out = (float*)d_out;

    const int B = in_sizes[0] / (SEQ * NI);   // 2048
    dim3 grid(B / 16), block(256);
    lstm_attn_fused_kernel<<<grid, block, 0, stream>>>(
        x, W_ih, W_hh, b_ih, b_hh, attn_w, fc1_w, fc1_b, fc2_w, fc2_b, out);
}
